// Linear_48120813585061
// MI455X (gfx1250) — matrix operand order repaired and checked by reading, not compiled, not run on hardware
//
#include <hip/hip_runtime.h>

// ---------- vector types ----------
typedef __attribute__((ext_vector_type(16))) __bf16 v16bf;
typedef __attribute__((ext_vector_type(8)))  float  v8f;
typedef __attribute__((ext_vector_type(8)))  __bf16 bf16x8;
typedef __attribute__((ext_vector_type(4)))  __bf16 bf16x4;

__device__ __forceinline__ __bf16 f2bf(float f) { return (__bf16)f; }

__device__ __forceinline__ v16bf cat8(bf16x8 lo, bf16x8 hi) {
  return __builtin_shufflevector(lo, hi, 0,1,2,3,4,5,6,7,8,9,10,11,12,13,14,15);
}

// Async global->LDS copy, 16B per lane, GVS mode:
//   LDS[vdst_lds_addr] = MEM[saddr + vaddr]   (tracked by ASYNCcnt)
__device__ __forceinline__ void async_cp_b128(unsigned lds_addr, unsigned goff,
                                              const void* base) {
  asm volatile("global_load_async_to_lds_b128 %0, %1, %2"
               :: "v"(lds_addr), "v"(goff), "s"(base) : "memory");
}
__device__ __forceinline__ void wait_async_le8() {
  asm volatile("s_wait_asynccnt 0x8" ::: "memory");
}
__device__ __forceinline__ void wait_async_le0() {
  asm volatile("s_wait_asynccnt 0x0" ::: "memory");
}

// =====================================================================
// Stage 0: W_enc f32 -> bf16 (one pass; lets the big GEMM stream bf16
// and use async b128 copies). Grid-stride over float4 chunks.
// =====================================================================
__global__ __launch_bounds__(256)
void convert_bf16(const float* __restrict__ src, __bf16* __restrict__ dst,
                  size_t n4) {
  size_t i = (size_t)blockIdx.x * 256 + threadIdx.x;
  const size_t stride = (size_t)gridDim.x * 256;
  for (; i < n4; i += stride) {
    float4 v = ((const float4*)src)[i];
    bf16x4 b = { f2bf(v.x), f2bf(v.y), f2bf(v.z), f2bf(v.w) };
    ((bf16x4*)dst)[i] = b;
  }
}

// =====================================================================
// Stage 1 (small, 17 GFLOP): sae_in(bf16) = x @ base_W^T + base_b - b_dec
// NT GEMM, f32 sources converted in regs, single-buffered LDS (as round 1).
// Block 128x128, BK=32, 8 waves (2x4), wave = 64x32 out.
// =====================================================================
__global__ __launch_bounds__(256)
void gemm_base(const float* __restrict__ Af, const float* __restrict__ Bmat,
               const float* __restrict__ add_vec, const float* __restrict__ sub_vec,
               __bf16* __restrict__ outB, int Ncols, int Kdim)
{
  constexpr int LDSS = 40;
  __shared__ __bf16 As[128 * LDSS];
  __shared__ __bf16 Bs[128 * LDSS];

  const int tid  = threadIdx.x;
  const int lane = tid & 31;
  const int wave = tid >> 5;
  const int wm = wave >> 2, wn = wave & 3;
  const int lm = lane & 15, lh = lane >> 4;
  const int m0 = blockIdx.x * 128;
  const int n0 = blockIdx.y * 128;

  v8f acc[4][2];
#pragma unroll
  for (int i = 0; i < 4; i++)
#pragma unroll
    for (int j = 0; j < 2; j++) acc[i][j] = v8f{};

  for (int k0 = 0; k0 < Kdim; k0 += 32) {
#pragma unroll
    for (int i = 0; i < 4; i++) {
      int linear = tid + i * 256;
      int r = linear >> 3, c4 = (linear & 7) * 4;
      float4 v = *(const float4*)(Af + (size_t)(m0 + r) * Kdim + k0 + c4);
      bf16x4 b = { f2bf(v.x), f2bf(v.y), f2bf(v.z), f2bf(v.w) };
      *(bf16x4*)(&As[r * LDSS + c4]) = b;
    }
#pragma unroll
    for (int i = 0; i < 4; i++) {
      int linear = tid + i * 256;
      int r = linear >> 3, c4 = (linear & 7) * 4;
      float4 v = *(const float4*)(Bmat + (size_t)(n0 + r) * Kdim + k0 + c4);
      bf16x4 b = { f2bf(v.x), f2bf(v.y), f2bf(v.z), f2bf(v.w) };
      *(bf16x4*)(&Bs[r * LDSS + c4]) = b;
    }
    __syncthreads();

    v16bf af[4], bf_[2];
    const int akb = lh ? 8 : 0;
#pragma unroll
    for (int t = 0; t < 4; t++) {
      int m = wm * 64 + t * 16 + lm;
      af[t] = cat8(*(const bf16x8*)(&As[m * LDSS + akb]),
                   *(const bf16x8*)(&As[m * LDSS + akb + 16]));
    }
    const int bkh = lh ? 8 : 0;
#pragma unroll
    for (int t = 0; t < 2; t++) {
      int n = wn * 32 + t * 16 + lm;
      bf_[t] = cat8(*(const bf16x8*)(&Bs[n * LDSS + bkh]),
                    *(const bf16x8*)(&Bs[n * LDSS + bkh + 16]));
    }
#pragma unroll
    for (int tm = 0; tm < 4; tm++)
#pragma unroll
      for (int tn = 0; tn < 2; tn++)
        acc[tm][tn] = __builtin_amdgcn_wmma_f32_16x16x32_bf16(
            false, af[tm], false, bf_[tn], (short)0, acc[tm][tn], false, false);
    __syncthreads();
  }

#pragma unroll
  for (int tn = 0; tn < 2; tn++) {
    const int col = n0 + wn * 32 + tn * 16 + lm;
    const float bias = add_vec[col] - sub_vec[col];
#pragma unroll
    for (int tm = 0; tm < 4; tm++) {
      const int rbase = m0 + wm * 64 + tm * 16 + (lh ? 8 : 0);
#pragma unroll
      for (int v = 0; v < 8; v++)
        outB[(size_t)(rbase + v) * Ncols + col] = f2bf(acc[tm][tn][v] + bias);
    }
  }
}

// =====================================================================
// Stage 2 (dominant, 550 GFLOP): pre_act(f32) = sae_in @ W_enc_bf16^T
// NT GEMM, both operands bf16. Block 128x128, BK=64 (2 WMMA K-substeps
// per staged tile), double-buffered LDS filled by async global->LDS
// b128 copies (ASYNCcnt); next tile's copies overlap the WMMA burst.
// LDS row stride 72 bf16 (144 B = 36 banks) -> conflict-free b128 frags.
// =====================================================================
__global__ __launch_bounds__(256)
void gemm_enc(const __bf16* __restrict__ A, const __bf16* __restrict__ B,
              float* __restrict__ out, int Ncols, int Kdim)
{
  constexpr int LDSS  = 72;             // bf16 elems per row
  constexpr int TILEE = 128 * LDSS;     // elems per buffer
  constexpr int TILEB = TILEE * 2;      // bytes per buffer
  __shared__ __bf16 As[2 * TILEE];
  __shared__ __bf16 Bs[2 * TILEE];

  const int tid  = threadIdx.x;
  const int lane = tid & 31;
  const int wave = tid >> 5;
  const int wm = wave >> 2, wn = wave & 3;
  const int lm = lane & 15, lh = lane >> 4;
  const int m0 = blockIdx.x * 128;      // x fastest -> consecutive blocks share B tile
  const int n0 = blockIdx.y * 128;

  // per-thread copy slots: 1024 chunks of 16B per (matrix, tile); 4 each
  unsigned a_lds[4], b_lds[4], a_goff[4], b_goff[4];
#pragma unroll
  for (int i = 0; i < 4; i++) {
    int c = i * 256 + tid;
    int r = c >> 3, c8 = (c & 7) * 8;
    a_lds[i]  = (unsigned)(uintptr_t)&As[r * LDSS + c8];
    b_lds[i]  = (unsigned)(uintptr_t)&Bs[r * LDSS + c8];
    a_goff[i] = (unsigned)(((m0 + r) * Kdim + c8) * 2);
    b_goff[i] = (unsigned)(((n0 + r) * Kdim + c8) * 2);
  }

  auto issue = [&](int kbyte, int buf) {
#pragma unroll
    for (int i = 0; i < 4; i++)
      async_cp_b128(a_lds[i] + buf * TILEB, a_goff[i] + kbyte, A);
#pragma unroll
    for (int i = 0; i < 4; i++)
      async_cp_b128(b_lds[i] + buf * TILEB, b_goff[i] + kbyte, B);
  };

  v8f acc[4][2];
#pragma unroll
  for (int i = 0; i < 4; i++)
#pragma unroll
    for (int j = 0; j < 2; j++) acc[i][j] = v8f{};

  const int nk = Kdim / 64;
  issue(0, 0);
  for (int kt = 0; kt < nk; kt++) {
    if (kt + 1 < nk) {                  // prefetch next tile into other buffer
      issue((kt + 1) * 128, (kt + 1) & 1);
      wait_async_le8();                 // oldest 8 (= this tile) complete
    } else {
      wait_async_le0();
    }
    __syncthreads();

    const __bf16* Ab = &As[(kt & 1) * TILEE];
    const __bf16* Bb = &Bs[(kt & 1) * TILEE];
#pragma unroll
    for (int s = 0; s < 2; s++) {       // two 32-K WMMA substeps
      const int kl = s * 32;
      v16bf af[4], bf_[2];
      const int akb = kl + (lh ? 8 : 0);
#pragma unroll
      for (int t = 0; t < 4; t++) {
        int m = wm * 64 + t * 16 + lm;
        af[t] = cat8(*(const bf16x8*)(Ab + m * LDSS + akb),
                     *(const bf16x8*)(Ab + m * LDSS + akb + 16));
      }
      const int bkh = kl + (lh ? 8 : 0);
#pragma unroll
      for (int t = 0; t < 2; t++) {
        int n = wn * 32 + t * 16 + lm;
        bf_[t] = cat8(*(const bf16x8*)(Bb + n * LDSS + bkh),
                      *(const bf16x8*)(Bb + n * LDSS + bkh + 16));
      }
#pragma unroll
      for (int tm = 0; tm < 4; tm++)
#pragma unroll
        for (int tn = 0; tn < 2; tn++)
          acc[tm][tn] = __builtin_amdgcn_wmma_f32_16x16x32_bf16(
              false, af[tm], false, bf_[tn], (short)0, acc[tm][tn], false, false);
    }
    __syncthreads();
  }

#pragma unroll
  for (int tn = 0; tn < 2; tn++) {
    const int col = n0 + wn * 32 + tn * 16 + lm;
#pragma unroll
    for (int tm = 0; tm < 4; tm++) {
      const int rbase = m0 + wm * 64 + tm * 16 + (lh ? 8 : 0);
#pragma unroll
      for (int v = 0; v < 8; v++)
        out[(size_t)(rbase + v) * Ncols + col] = acc[tm][tn][v];
    }
  }
}

// =====================================================================
// Stage 3: per-row top-32 over L values. One block (256 threads) per row.
// =====================================================================
__global__ __launch_bounds__(256)
void topk_kernel(const float* __restrict__ pre, float* __restrict__ tv,
                 int* __restrict__ ti, int L)
{
  const int row = blockIdx.x;
  const int tid = threadIdx.x;
  const float* p = pre + (size_t)row * L;

  float lv[32]; int li[32];
#pragma unroll
  for (int i = 0; i < 32; i++) { lv[i] = -3.0e38f; li[i] = -1; }

  for (int c = tid; c < L; c += 256) {
    float v = p[c];
    if (v > lv[31]) {
      int j = 31;
      while (j > 0 && lv[j - 1] < v) { lv[j] = lv[j - 1]; li[j] = li[j - 1]; j--; }
      lv[j] = v; li[j] = c;
    }
  }

  __shared__ float sv[256];
  __shared__ int   si[256];
  int ptr = 0;
  for (int r = 0; r < 32; r++) {
    sv[tid] = (ptr < 32) ? lv[ptr] : -3.0e38f;
    si[tid] = (ptr < 32) ? li[ptr] : -1;
    __syncthreads();
    for (int s = 128; s > 0; s >>= 1) {
      if (tid < s && sv[tid + s] > sv[tid]) { sv[tid] = sv[tid + s]; si[tid] = si[tid + s]; }
      __syncthreads();
    }
    float mv = sv[0]; int mi = si[0];
    __syncthreads();
    if (ptr < 32 && li[ptr] == mi && mi >= 0) ptr++;   // unique winner advances
    if (tid == 0) { tv[row * 32 + r] = mv; ti[row * 32 + r] = mi; }
  }
}

// =====================================================================
// Stage 4: out[row,:] = sum_j tv[row,j] * W_dec[ti[row,j],:] + b_dec
// =====================================================================
__global__ __launch_bounds__(256)
void decode_kernel(const float* __restrict__ tv, const int* __restrict__ ti,
                   const float* __restrict__ Wdec, const float* __restrict__ bdec,
                   float* __restrict__ out, int D)
{
  const int row = blockIdx.x;
  const int tid = threadIdx.x;
  __shared__ float svv[32];
  __shared__ int   sii[32];
  if (tid < 32) { svv[tid] = tv[row * 32 + tid]; sii[tid] = ti[row * 32 + tid]; }
  __syncthreads();

  float acc[8];
#pragma unroll
  for (int i = 0; i < 8; i++) acc[i] = 0.0f;

  for (int j = 0; j < 32; j++) {
    const float* w = Wdec + (size_t)sii[j] * D;
    const float s = svv[j];
#pragma unroll
    for (int i = 0; i < 8; i++) acc[i] += s * w[tid + i * 256];
  }
#pragma unroll
  for (int i = 0; i < 8; i++)
    out[(size_t)row * D + tid + i * 256] = acc[i] + bdec[tid + i * 256];
}

// =====================================================================
extern "C" void kernel_launch(void* const* d_in, const int* in_sizes, int n_in,
                              void* d_out, int out_size, void* d_ws, size_t ws_size,
                              hipStream_t stream) {
  const float* x     = (const float*)d_in[0];
  const float* baseW = (const float*)d_in[1];
  const float* baseb = (const float*)d_in[2];
  const float* Wenc  = (const float*)d_in[3];
  const float* Wdec  = (const float*)d_in[4];
  const float* bdec  = (const float*)d_in[5];

  const int D = in_sizes[2];          // 2048
  const int N = in_sizes[0] / D;      // 2048
  const int L = in_sizes[3] / D;      // 65536

  char* ws = (char*)d_ws;
  __bf16* sae_in = (__bf16*)ws;                       // N*D bf16   (8 MB)
  size_t off = ((size_t)N * D * 2 + 255) & ~(size_t)255;
  __bf16* Wenc_bf = (__bf16*)(ws + off);              // L*D bf16   (256 MB)
  off += (size_t)L * D * 2;
  float* pre = (float*)(ws + off);                    // N*L f32    (512 MB)
  off += (size_t)N * L * 4;
  float* tv = (float*)(ws + off);  off += (size_t)N * 32 * 4;
  int*   ti = (int*)(ws + off);

  dim3 blk(256);
  // stage 0: W_enc -> bf16
  convert_bf16<<<dim3(16384), blk, 0, stream>>>(Wenc, Wenc_bf, (size_t)L * D / 4);
  // stage 1: sae_in(bf16) = x @ base_W^T + base_b - b_dec
  gemm_base<<<dim3(N / 128, D / 128), blk, 0, stream>>>(x, baseW, baseb, bdec,
                                                        sae_in, D, D);
  // stage 2: pre_act(f32) = sae_in @ W_enc^T  (async-LDS double-buffered WMMA)
  gemm_enc<<<dim3(N / 128, L / 128), blk, 0, stream>>>(sae_in, Wenc_bf, pre, L, D);
  // stage 3: per-row top-32
  topk_kernel<<<dim3(N), blk, 0, stream>>>(pre, tv, ti, L);
  // stage 4: decode gather-weighted sum + b_dec
  decode_kernel<<<dim3(N), blk, 0, stream>>>(tv, ti, Wdec, bdec, (float*)d_out, D);
}
